// EchoMultiHeadAttention_16080357556564
// MI455X (gfx1250) — matrix operand order repaired and checked by reading, not compiled, not run on hardware
//
#include <hip/hip_runtime.h>

// Problem constants (match reference)
#define BB 2
#define SS 2048
#define DD 1024
#define HH 16
#define HD 64
#define MM (BB * SS) // 4096 rows of the [B*S, D] activation matrix

typedef __attribute__((ext_vector_type(16))) _Float16 v16h;
typedef __attribute__((ext_vector_type(4)))  _Float16 v4h;
typedef __attribute__((ext_vector_type(8)))  float    v8f;

union V16 {
    v16h v;
    uint4 u[2];
};

// ---- async global -> LDS copy (CDNA5 GLOBAL_LOAD_ASYNC_TO_LDS_B128) -------
// Inline asm: bypasses the clang builtin's CUDA-address-space parameter types.
#if defined(__gfx1250__)
#define USE_ASYNC_LDS 1
#else
#define USE_ASYNC_LDS 0
#endif

#if USE_ASYNC_LDS
__device__ __forceinline__ void async_copy_b128(const void* g, void* l) {
    // LDS operand = low 32 bits of the generic LDS address (ISA 10.2);
    // global operand = 64-bit VGPR pair.
    unsigned lds = (unsigned)(uintptr_t)l;
    unsigned long long ga = (unsigned long long)(uintptr_t)g;
    asm volatile("global_load_async_to_lds_b128 %0, %1, off"
                 :: "v"(lds), "v"(ga)
                 : "memory");
}
#endif

__device__ __forceinline__ void wait_async0() {
#if __has_builtin(__builtin_amdgcn_s_wait_asynccnt)
    __builtin_amdgcn_s_wait_asynccnt(0);
#else
    asm volatile("s_wait_asynccnt 0x0" ::: "memory");
#endif
}

// ---- WMMA operand loaders -------------------------------------------------
// A operand (16x32 f16, rows = M): lane (r = lane&15, h = lane>>4) holds row r.
//   elements 0..7  <- K = k0 + 8h .. +7      (16B)
//   elements 8..15 <- K = k0 + 16 + 8h .. +7 (16B)
__device__ __forceinline__ v16h load_a16(const _Float16* rowBase, int k0, int lh) {
    V16 t;
    t.u[0] = *(const uint4*)(rowBase + k0 + 8 * lh);
    t.u[1] = *(const uint4*)(rowBase + k0 + 16 + 8 * lh);
    return t.v;
}

// B operand (32x16 f16, cols = N): lane (r, h) holds column r.
//   elements 0..7  <- K = k0 + 8h .. +7      (16B)
//   elements 8..15 <- K = k0 + 16 + 8h .. +7 (16B)
__device__ __forceinline__ v16h load_b16(const _Float16* rowBase, int k0, int lh) {
    V16 t;
    const _Float16* p = rowBase + k0 + 8 * lh;
    t.u[0] = *(const uint4*)(p);
    t.u[1] = *(const uint4*)(p + 16);
    return t.v;
}

// ---- 16-lane butterfly reductions via v_permlane16_b32 (VALU, no LDS) -----
__device__ __forceinline__ float perm16(float x, int lo, int hi) {
    int v = __float_as_int(x);
    int y = __builtin_amdgcn_permlane16(v, v, lo, hi, false, false);
    return __int_as_float(y);
}
__device__ __forceinline__ float bfly16_max(float x) {
    x = fmaxf(x, perm16(x, 0x67452301, 0xEFCDAB89)); // xor 1
    x = fmaxf(x, perm16(x, 0x54761032, 0xDCFE98BA)); // xor 2
    x = fmaxf(x, perm16(x, 0x32107654, 0xBA98FEDC)); // xor 4
    x = fmaxf(x, perm16(x, 0xFEDCBA98, 0x76543210)); // xor 8
    return x;
}
__device__ __forceinline__ float bfly16_sum(float x) {
    x += perm16(x, 0x67452301, 0xEFCDAB89); // xor 1
    x += perm16(x, 0x54761032, 0xDCFE98BA); // xor 2
    x += perm16(x, 0x32107654, 0xBA98FEDC); // xor 4
    x += perm16(x, 0xFEDCBA98, 0x76543210); // xor 8
    return x;
}

// Wave-private LDS ordering: compiler fence + scheduling barrier. Hardware
// keeps a wave's LDS ops in order; compiler inserts s_wait_dscnt for uses.
__device__ __forceinline__ void wave_lds_fence() {
    __asm__ volatile("" ::: "memory");
    __builtin_amdgcn_wave_barrier();
    __asm__ volatile("" ::: "memory");
}

// ---- fp32 -> f16 conversion ----------------------------------------------
__global__ __launch_bounds__(256) void cvt_f32_f16(_Float16* __restrict__ dst,
                                                   const float* __restrict__ src,
                                                   int n4) {
    int i = blockIdx.x * blockDim.x + threadIdx.x;
    int stride = gridDim.x * blockDim.x;
    for (; i < n4; i += stride) {
        float4 f = ((const float4*)src)[i];
        v4h h;
        h[0] = (_Float16)f.x; h[1] = (_Float16)f.y;
        h[2] = (_Float16)f.z; h[3] = (_Float16)f.w;
        ((v4h*)dst)[i] = h;
    }
}

// ---- WMMA GEMM: C[m,n] = A[m,:] . Bw[n,:] + bias[n] -----------------------
// MODE 0: f16 out, scatter to [b,h,s,d]       (Q, K projections)
// MODE 1: f16 out, scatter to [b,h,d,s]       (V projection, transposed)
// MODE 2: f32 out, row-major [M,N]            (output projection -> d_out)
template <int MODE>
__global__ __launch_bounds__(128) void gemm_wmma(const _Float16* __restrict__ A,
                                                 const _Float16* __restrict__ Bw,
                                                 const float* __restrict__ bias,
                                                 void* __restrict__ out,
                                                 int N, int K) {
    __shared__ _Float16 ldsB[2][64 * 32]; // double-buffered 64 n-rows x 32 k

    const int tid   = threadIdx.x;
    const int w     = tid >> 5;
    const int lane  = tid & 31;
    const int laneR = lane & 15;
    const int lh    = lane >> 4;
    const int nBase = blockIdx.x * 64;
    const int mBase = blockIdx.y * 64;
    const int aRow  = mBase + w * 16 + laneR;
    const _Float16* aPtr = A + (size_t)aRow * K;

    const int nl = tid >> 1;
    const int ch = (tid & 1) * 16;
    const _Float16* bSrc = Bw + (size_t)(nBase + nl) * K + ch;
    _Float16* dst0 = &ldsB[0][nl * 32 + ch];
    _Float16* dst1 = &ldsB[1][nl * 32 + ch];

    v8f acc[4] = {};

#if USE_ASYNC_LDS
    // prologue: async-stage slab 0 (no VGPR round trip, tracked by ASYNCcnt)
    async_copy_b128(bSrc, dst0);
    async_copy_b128(bSrc + 8, dst0 + 8);
    for (int k0 = 0; k0 < K; k0 += 32) {
        const int buf = (k0 >> 5) & 1;
        wait_async0();       // own async stores to LDS done
        __syncthreads();     // -> whole slab visible, prev buffer free
        if (k0 + 32 < K) {   // async-stage next slab into the other buffer
            _Float16* nd = buf ? dst0 : dst1;
            async_copy_b128(bSrc + k0 + 32, nd);
            async_copy_b128(bSrc + k0 + 40, nd + 8);
        }
        v16h a = load_a16(aPtr, k0, lh);
#pragma unroll
        for (int nt = 0; nt < 4; ++nt) {
            v16h b = load_b16(&ldsB[buf][(nt * 16 + laneR) * 32], 0, lh);
            acc[nt] = __builtin_amdgcn_wmma_f32_16x16x32_f16(
                false, a, false, b, (short)0, acc[nt], false, false);
        }
    }
#else
    // fallback: register double-buffer + alternating LDS buffers
    uint4 r0 = *(const uint4*)(bSrc);
    uint4 r1 = *(const uint4*)(bSrc + 8);
    for (int k0 = 0; k0 < K; k0 += 32) {
        const int buf = (k0 >> 5) & 1;
        _Float16* d = buf ? dst1 : dst0;
        *(uint4*)(d)     = r0;
        *(uint4*)(d + 8) = r1;
        __syncthreads();
        if (k0 + 32 < K) {
            r0 = *(const uint4*)(bSrc + k0 + 32);
            r1 = *(const uint4*)(bSrc + k0 + 40);
        }
        v16h a = load_a16(aPtr, k0, lh);
#pragma unroll
        for (int nt = 0; nt < 4; ++nt) {
            v16h b = load_b16(&ldsB[buf][(nt * 16 + laneR) * 32], 0, lh);
            acc[nt] = __builtin_amdgcn_wmma_f32_16x16x32_f16(
                false, a, false, b, (short)0, acc[nt], false, false);
        }
    }
#endif

    // Epilogue: C element j -> (row = mBase+w*16+j+8*lh, col = nBase+nt*16+laneR)
#pragma unroll
    for (int nt = 0; nt < 4; ++nt) {
        const int ncol = nBase + nt * 16 + laneR;
        const float bb = bias[ncol];
#pragma unroll
        for (int j = 0; j < 8; ++j) {
            const int mrow = mBase + w * 16 + j + 8 * lh;
            const float val = acc[nt][j] + bb;
            if (MODE == 2) {
                ((float*)out)[(size_t)mrow * N + ncol] = val;
            } else {
                const int bIdx = mrow >> 11;       // / S
                const int s    = mrow & (SS - 1);
                const int head = ncol >> 6;        // / HD
                const int d    = ncol & (HD - 1);
                if (MODE == 0)
                    ((_Float16*)out)[(((size_t)(bIdx * HH + head)) * SS + s) * HD + d] =
                        (_Float16)val;
                else
                    ((_Float16*)out)[(((size_t)(bIdx * HH + head)) * HD + d) * SS + s] =
                        (_Float16)val;
            }
        }
    }
}

// ---- Flash attention: per (b, head, 64-q block); 4 waves, 16 q-rows each --
__global__ __launch_bounds__(128) void attn_wmma(const _Float16* __restrict__ Q,
                                                 const _Float16* __restrict__ Km,
                                                 const _Float16* __restrict__ Vt,
                                                 const float* __restrict__ mask,
                                                 _Float16* __restrict__ O) {
    __shared__ _Float16 ldsP[4][16 * 32]; // per-wave P transpose staging, 4 KB

    const int tid   = threadIdx.x;
    const int w     = tid >> 5;
    const int lane  = tid & 31;
    const int laneR = lane & 15;
    const int lh    = lane >> 4;
    const int b     = blockIdx.z;
    const int head  = blockIdx.y;
    const int q0    = blockIdx.x * 64 + w * 16;

    const size_t bh = (size_t)(b * HH + head);
    const _Float16* qBase = Q + bh * SS * HD;
    const _Float16* kBase = Km + bh * SS * HD;
    const _Float16* vBase = Vt + bh * HD * SS;
    const float*    mBase = mask + (size_t)b * SS * SS;

    // Q strip held in registers as two A operands (d = 0..31, 32..63)
    const _Float16* qRow = qBase + (size_t)(q0 + laneR) * HD;
    const v16h aq0 = load_a16(qRow, 0, lh);
    const v16h aq1 = load_a16(qRow, 32, lh);

    float mrun[8], lrun[8];
#pragma unroll
    for (int j = 0; j < 8; ++j) { mrun[j] = -1e30f; lrun[j] = 0.f; }
    v8f oacc[4] = {};
    const float scale = 0.125f; // HD^-0.5

    for (int ks = 0; ks < SS; ks += 32) {
        // --- scores for 32 key columns: two 16-wide C tiles
        const _Float16* kRowA = kBase + (size_t)(ks + laneR) * HD;
        const _Float16* kRowB = kBase + (size_t)(ks + 16 + laneR) * HD;
        v16h bkA0 = load_b16(kRowA, 0, lh);
        v16h bkA1 = load_b16(kRowA, 32, lh);
        v16h bkB0 = load_b16(kRowB, 0, lh);
        v16h bkB1 = load_b16(kRowB, 32, lh);
        v8f z = {};
        v8f scA = __builtin_amdgcn_wmma_f32_16x16x32_f16(false, aq0, false, bkA0,
                                                         (short)0, z, false, false);
        scA = __builtin_amdgcn_wmma_f32_16x16x32_f16(false, aq1, false, bkA1,
                                                     (short)0, scA, false, false);
        v8f scB = __builtin_amdgcn_wmma_f32_16x16x32_f16(false, aq0, false, bkB0,
                                                         (short)0, z, false, false);
        scB = __builtin_amdgcn_wmma_f32_16x16x32_f16(false, aq1, false, bkB1,
                                                     (short)0, scB, false, false);

        // prefetch next iteration's K/V rows into near cache (L2 is resident)
        if (ks + 32 < SS) {
            __builtin_prefetch(kBase + (size_t)(ks + 32 + laneR) * HD, 0, 3);
            __builtin_prefetch(vBase + (size_t)laneR * SS + ks + 32, 0, 3);
        }

        // --- online softmax; each C row lives in one 16-lane group
#pragma unroll
        for (int j = 0; j < 8; ++j) {
            const int qg = q0 + j + 8 * lh;
            const float* mr = mBase + (size_t)qg * SS + ks;
            float sa = scA[j] * scale + mr[laneR];
            float sb = scB[j] * scale + mr[16 + laneR];
            float tmax = bfly16_max(fmaxf(sa, sb));
            const float mn   = fmaxf(mrun[j], tmax);
            const float resc = __expf(mrun[j] - mn);
            const float pa   = __expf(sa - mn);
            const float pb   = __expf(sb - mn);
            const float psum = bfly16_sum(pa + pb);
            lrun[j] = lrun[j] * resc + psum;
            mrun[j] = mn;
#pragma unroll
            for (int t = 0; t < 4; ++t) oacc[t][j] = oacc[t][j] * resc;
            // store P (C layout -> row-major 16x32 tile in LDS)
            ldsP[w][(j + 8 * lh) * 32 + laneR]      = (_Float16)pa;
            ldsP[w][(j + 8 * lh) * 32 + 16 + laneR] = (_Float16)pb;
        }
        wave_lds_fence(); // wave-private tile: no cross-wave barrier needed

        // --- P back as A operand; PV WMMAs against transposed V rows
        v16h ap = load_a16(&ldsP[w][laneR * 32], 0, lh);
#pragma unroll
        for (int t = 0; t < 4; ++t) {
            const _Float16* vRow = vBase + (size_t)(t * 16 + laneR) * SS;
            v16h bvv = load_b16(vRow, ks, lh);
            oacc[t] = __builtin_amdgcn_wmma_f32_16x16x32_f16(
                false, ap, false, bvv, (short)0, oacc[t], false, false);
        }
        wave_lds_fence();
    }

    // --- normalize and scatter to o16 [b, s, head*64 + d] (f16)
#pragma unroll
    for (int j = 0; j < 8; ++j) {
        const float inv = 1.f / lrun[j];
        const int qg = q0 + j + 8 * lh;
        _Float16* orow = O + ((size_t)(b * SS + qg)) * DD + head * HD;
#pragma unroll
        for (int t = 0; t < 4; ++t)
            orow[t * 16 + laneR] = (_Float16)(oacc[t][j] * inv);
    }
}

extern "C" void kernel_launch(void* const* d_in, const int* in_sizes, int n_in,
                              void* d_out, int out_size, void* d_ws, size_t ws_size,
                              hipStream_t stream) {
    const float* hs   = (const float*)d_in[0];
    const float* mask = (const float*)d_in[1];
    const float* Wq   = (const float*)d_in[2];
    const float* bq   = (const float*)d_in[3];
    const float* Wk   = (const float*)d_in[4];
    const float* bk   = (const float*)d_in[5];
    const float* Wv   = (const float*)d_in[6];
    const float* bv   = (const float*)d_in[7];
    const float* Wo   = (const float*)d_in[8];
    const float* bo   = (const float*)d_in[9];

    char* ws = (char*)d_ws;
    size_t off = 0;
    auto carve = [&](size_t bytes) -> void* {
        void* p = ws + off;
        off += (bytes + 255) & ~(size_t)255;
        return p;
    };
    _Float16* hs16 = (_Float16*)carve((size_t)MM * DD * 2); // 8 MB
    _Float16* wq16 = (_Float16*)carve((size_t)DD * DD * 2); // 2 MB
    _Float16* wk16 = (_Float16*)carve((size_t)DD * DD * 2);
    _Float16* wv16 = (_Float16*)carve((size_t)DD * DD * 2);
    _Float16* wo16 = (_Float16*)carve((size_t)DD * DD * 2);
    _Float16* q16  = (_Float16*)carve((size_t)MM * DD * 2); // [b,h,s,d]
    _Float16* k16  = (_Float16*)carve((size_t)MM * DD * 2); // [b,h,s,d]
    _Float16* vt16 = (_Float16*)carve((size_t)MM * DD * 2); // [b,h,d,s]
    _Float16* o16  = (_Float16*)carve((size_t)MM * DD * 2); // [b*s, d]
    (void)in_sizes; (void)n_in; (void)out_size; (void)ws_size;

    // fp32 -> f16
    cvt_f32_f16<<<dim3(1024), dim3(256), 0, stream>>>(hs16, hs, MM * DD / 4);
    cvt_f32_f16<<<dim3(256), dim3(256), 0, stream>>>(wq16, Wq, DD * DD / 4);
    cvt_f32_f16<<<dim3(256), dim3(256), 0, stream>>>(wk16, Wk, DD * DD / 4);
    cvt_f32_f16<<<dim3(256), dim3(256), 0, stream>>>(wv16, Wv, DD * DD / 4);
    cvt_f32_f16<<<dim3(256), dim3(256), 0, stream>>>(wo16, Wo, DD * DD / 4);

    // Q/K/V projections
    dim3 gblk(128), ggrd(DD / 64, MM / 64);
    gemm_wmma<0><<<ggrd, gblk, 0, stream>>>(hs16, wq16, bq, (void*)q16, DD, DD);
    gemm_wmma<0><<<ggrd, gblk, 0, stream>>>(hs16, wk16, bk, (void*)k16, DD, DD);
    gemm_wmma<1><<<ggrd, gblk, 0, stream>>>(hs16, wv16, bv, (void*)vt16, DD, DD);

    // fused attention
    attn_wmma<<<dim3(SS / 64, HH, BB), dim3(128), 0, stream>>>(q16, k16, vt16, mask, o16);

    // output projection -> fp32 d_out
    gemm_wmma<2><<<ggrd, gblk, 0, stream>>>(o16, wo16, bo, d_out, DD, DD);
}
